// SS2DLBlock_31190052503715
// MI455X (gfx1250) — matrix operand order repaired and checked by reading, not compiled, not run on hardware
//
#include <hip/hip_runtime.h>

typedef __attribute__((ext_vector_type(16))) _Float16 v16h;
typedef __attribute__((ext_vector_type(8)))  _Float16 v8h;
typedef __attribute__((ext_vector_type(8)))  float    v8f;

#define Bn   4
#define Cn   128
#define Hn   48
#define Wn   48
#define Ln   (Hn*Wn)      /* 2304 */
#define DIn  256
#define Kdir 4
#define Nst  16
#define Rr   8

__device__ __forceinline__ float siluf(float x)     { return x / (1.f + __expf(-x)); }
__device__ __forceinline__ float softplusf(float x) { return (x > 20.f) ? x : log1pf(__expf(x)); }

// ---------------------------------------------------------------------------
// f16 fragment loaders (no guards, pure b128 traffic).
// A-frag (16-bit A 16x32): lane holds row M=lane&15; elements e -> K = kb + 16*hi + e.
// B-frag: lane holds col N=lane&15, elements
// e -> K = kb + 16*hi + e (16 contiguous halves).
// ---------------------------------------------------------------------------
__device__ __forceinline__ v16h load_afrag(const _Float16* __restrict__ Arow, int kb, int hi)
{
    return *(const v16h*)(Arow + kb + 16 * hi);
}
__device__ __forceinline__ v16h load_bfrag(const _Float16* __restrict__ Wrow, int kb, int hi)
{
    return *(const v16h*)(Wrow + kb + 16 * hi);
}

// ---------------------------------------------------------------------------
// WMMA GEMM: D[M,N] = epi( A[M,K](f16) * W[N,K](f16)^T ).
// Requirements: M%64==0 (4 waves/block over M), N%(16*NT)==0, K%32==0.
// One wave computes a 16 x (16*NT) strip; A fragment reused across NT B tiles.
// ---------------------------------------------------------------------------
template<int NT>
__global__ void gemm_wmma_kernel(const _Float16* __restrict__ A, const _Float16* __restrict__ Wm,
                                 const float* __restrict__ bias, const float* __restrict__ resid,
                                 float* __restrict__ Dout, _Float16* __restrict__ Dout16,
                                 int M, int N, int Kd, int act, float slope,
                                 int resid_mode, int out_half)
{
    const int wv   = threadIdx.x >> 5;
    const int lane = threadIdx.x & 31;
    const int hi   = lane >> 4;
    const int lo   = lane & 15;
    const int mtile = blockIdx.y * 4 + wv;
    const int nbase = blockIdx.x * (16 * NT);
    const _Float16* Arow = A + (size_t)(mtile * 16 + lo) * Kd;
    const _Float16* Wr[NT];
#pragma unroll
    for (int t = 0; t < NT; ++t) Wr[t] = Wm + (size_t)(nbase + t * 16 + lo) * Kd;
    v8f acc[NT];
#pragma unroll
    for (int t = 0; t < NT; ++t) acc[t] = (v8f){};
    for (int kb = 0; kb < Kd; kb += 32) {
        v16h af = load_afrag(Arow, kb, hi);
#pragma unroll
        for (int t = 0; t < NT; ++t) {
            v16h bf = load_bfrag(Wr[t], kb, hi);
            acc[t] = __builtin_amdgcn_wmma_f32_16x16x32_f16(false, af, false, bf,
                                                            (short)0, acc[t], false, false);
        }
    }
#pragma unroll
    for (int t = 0; t < NT; ++t) {
#pragma unroll
        for (int r = 0; r < 8; ++r) {
            int m = mtile * 16 + r + 8 * hi;
            int n = nbase + t * 16 + lo;
            float v = acc[t][r];
            if (bias) v += bias[n];
            if (act == 1) v = (v > 0.f) ? v : slope * v;
            if (resid_mode == 1) {
                v += resid[(size_t)m * N + n];
            } else if (resid_mode == 2) {              // residual stored (B,C,H,W)
                int b = m / Ln, l = m - b * Ln;
                v += resid[((size_t)(b * Cn + n)) * Ln + l];
            }
            if (out_half) Dout16[(size_t)m * N + n] = (_Float16)v;
            else          Dout [(size_t)m * N + n] = v;
        }
    }
}

// ---------------------------------------------------------------------------
// x_proj GEMM with padded weights (4 x 48 x 256 f16). Per (b,k):
// dbl[b,k,s,0..39] = xmc16[b, pix(s,k), :] * Wk^T. NT=3 tiles of N.
// grid: (1, Ln/64, Bn*Kdir), block 128.
// ---------------------------------------------------------------------------
__global__ void xproj_wmma_kernel(const _Float16* __restrict__ xmc16,
                                  const _Float16* __restrict__ wpad,
                                  float* __restrict__ dbl)
{
    const int wv   = threadIdx.x >> 5;
    const int lane = threadIdx.x & 31;
    const int hi   = lane >> 4;
    const int lo   = lane & 15;
    const int bk = blockIdx.z;
    const int k  = bk & 3;
    const int b  = bk >> 2;
    const bool colo = (k & 1);
    const int mtile = blockIdx.y * 4 + wv;
    const int s_row = mtile * 16 + lo;
    int pix = colo ? ((s_row % Hn) * Wn + (s_row / Hn)) : s_row;
    const _Float16* Arow = xmc16 + ((size_t)(b * Ln + pix)) * DIn;
    const _Float16* Wk   = wpad + (size_t)k * 48 * DIn;
    v8f acc[3];
#pragma unroll
    for (int t = 0; t < 3; ++t) acc[t] = (v8f){};
    for (int kb = 0; kb < DIn; kb += 32) {
        v16h af = load_afrag(Arow, kb, hi);
#pragma unroll
        for (int t = 0; t < 3; ++t) {
            v16h bf = load_bfrag(Wk + (size_t)(t * 16 + lo) * DIn, kb, hi);
            acc[t] = __builtin_amdgcn_wmma_f32_16x16x32_f16(false, af, false, bf,
                                                            (short)0, acc[t], false, false);
        }
    }
    float* dbase = dbl + (size_t)bk * Ln * 40;
#pragma unroll
    for (int t = 0; t < 3; ++t) {
#pragma unroll
        for (int r = 0; r < 8; ++r) {
            int s = mtile * 16 + r + 8 * hi;
            int n = t * 16 + lo;
            if (n < 40) dbase[(size_t)s * 40 + n] = acc[t][r];
        }
    }
}

// ------------------------- weight prep (once per launch) -------------------
__global__ void cvt16_kernel(const float* __restrict__ src, _Float16* __restrict__ dst, int n)
{
    int i = blockIdx.x * 256 + threadIdx.x;
    if (i < n) dst[i] = (_Float16)src[i];
}
__global__ void padw_kernel(const float* __restrict__ xpw, _Float16* __restrict__ wpad)
{
    int idx = blockIdx.x * 256 + threadIdx.x;     // Kdir*48*DIn
    if (idx >= Kdir * 48 * DIn) return;
    int c = idx % DIn; int rest = idx / DIn;
    int row = rest % 48; int k = rest / 48;
    wpad[idx] = (row < 40) ? (_Float16)xpw[((size_t)(k * 40 + row)) * DIn + c] : (_Float16)0.f;
}
__global__ void padc3_kernel(const float* __restrict__ w, _Float16* __restrict__ dst)
{
    int idx = blockIdx.x * 256 + threadIdx.x;     // Cn*32
    if (idx >= Cn * 32) return;
    int kk = idx % 32; int co = idx / 32;
    dst[idx] = (kk < 16) ? (_Float16)w[co * 16 + kk] : (_Float16)0.f;
}

// ---------------------------------------------------------------------------
// LayerNorm over C, input (B,C,H,W), output f16 rows (B*L, C) (feeds GEMMs only)
// ---------------------------------------------------------------------------
__global__ void ln_bchw_kernel(const float* __restrict__ x, const float* __restrict__ g,
                               const float* __restrict__ be, _Float16* __restrict__ out, float eps)
{
    __shared__ float red[Cn];
    int bl = blockIdx.x; int b = bl / Ln; int l = bl - b * Ln;
    int c = threadIdx.x;
    float v = x[((size_t)(b * Cn + c)) * Ln + l];
    red[c] = v; __syncthreads();
    for (int s = Cn / 2; s > 0; s >>= 1) { if (c < s) red[c] += red[c + s]; __syncthreads(); }
    float mean = red[0] / (float)Cn; __syncthreads();
    float dv = v - mean; red[c] = dv * dv; __syncthreads();
    for (int s = Cn / 2; s > 0; s >>= 1) { if (c < s) red[c] += red[c + s]; __syncthreads(); }
    float var = red[0] / (float)Cn;
    out[(size_t)bl * Cn + c] = (_Float16)(dv * rsqrtf(var + eps) * g[c] + be[c]);
}

// LayerNorm over C, f32 row input -> f16 row output (feeds fc1)
__global__ void ln_rows_kernel(const float* __restrict__ xin, const float* __restrict__ g,
                               const float* __restrict__ be, _Float16* __restrict__ out, float eps)
{
    __shared__ float red[Cn];
    int bl = blockIdx.x;
    int c = threadIdx.x;
    float v = xin[(size_t)bl * Cn + c];
    red[c] = v; __syncthreads();
    for (int s = Cn / 2; s > 0; s >>= 1) { if (c < s) red[c] += red[c + s]; __syncthreads(); }
    float mean = red[0] / (float)Cn; __syncthreads();
    float dv = v - mean; red[c] = dv * dv; __syncthreads();
    for (int s = Cn / 2; s > 0; s >>= 1) { if (c < s) red[c] += red[c + s]; __syncthreads(); }
    float var = red[0] / (float)Cn;
    out[(size_t)bl * Cn + c] = (_Float16)(dv * rsqrtf(var + eps) * g[c] + be[c]);
}

// ---------------------------------------------------------------------------
// Depthwise 3x3 + SiLU. xm = cols [0,256) of xz rows (stride 512).
// Writes f32 (for scan) and f16 (for x_proj GEMM).
// ---------------------------------------------------------------------------
__global__ void dwconv_kernel(const float* __restrict__ xz, const float* __restrict__ cw,
                              const float* __restrict__ cb, float* __restrict__ xmc,
                              _Float16* __restrict__ xmc16)
{
    int idx = blockIdx.x * 256 + threadIdx.x;
    if (idx >= Bn * Ln * DIn) return;
    int d = idx % DIn; int bl = idx / DIn;
    int b = bl / Ln; int l = bl - b * Ln;
    int hh = l / Wn; int ww = l - hh * Wn;
    float acc = cb[d];
#pragma unroll
    for (int ky = 0; ky < 3; ++ky) {
        int y = hh + ky - 1; if (y < 0 || y >= Hn) continue;
#pragma unroll
        for (int kx = 0; kx < 3; ++kx) {
            int x2 = ww + kx - 1; if (x2 < 0 || x2 >= Wn) continue;
            acc += cw[d * 9 + ky * 3 + kx]
                 * xz[((size_t)(b * Ln + y * Wn + x2)) * (2 * DIn) + d];
        }
    }
    float s = siluf(acc);
    xmc[(size_t)idx]   = s;
    xmc16[(size_t)idx] = (_Float16)s;
}

__global__ void zero_kernel(float* p, int n)
{
    int i = blockIdx.x * 256 + threadIdx.x;
    if (i < n) p[i] = 0.f;
}

// ---------------------------------------------------------------------------
// Selective scan. 4 waves/block; wave = 2 channels x 16 states (lane = dp*16+n).
// dt-projection (R=8) + softplus fused. Reversed directions iterate t backwards;
// storage index maps to the final row-major pixel for every direction, so all 4
// directions atomicAdd into one accumulator (includes D*u skip term).
// ---------------------------------------------------------------------------
__global__ void scan_kernel(const float* __restrict__ xmc, const float* __restrict__ dbl,
                            const float* __restrict__ dtw, const float* __restrict__ dtb,
                            const float* __restrict__ Alogs, const float* __restrict__ Dsv,
                            float* __restrict__ yc)
{
    int wv   = threadIdx.x >> 5;
    int lane = threadIdx.x & 31;
    int idx  = blockIdx.x * 4 + wv;        // over Bn*Kdir*(DIn/2) = 2048
    int dgrp = idx % (DIn / 2);
    int bk   = idx / (DIn / 2);
    int k = bk % Kdir; int b = bk / Kdir;
    int n = lane & 15; int dp = lane >> 4;
    int d  = dgrp * 2 + dp;
    int kd = k * DIn + d;
    float Ac = -__expf(Alogs[(size_t)kd * Nst + n]);
    float Dc = Dsv[kd];
    float wdt[Rr];
#pragma unroll
    for (int r = 0; r < Rr; ++r) wdt[r] = dtw[(size_t)kd * Rr + r];
    float bdt = dtb[kd];
    const bool rev  = (k >= 2);
    const bool colo = (k & 1);
    const float* dblk = dbl + (size_t)bk * Ln * 40;
    float h = 0.f;
    for (int t = 0; t < Ln; ++t) {
        int s = rev ? (Ln - 1 - t) : t;
        int pix;
        if (colo) { int hh = s % Hn; int ww = s / Hn; pix = hh * Wn + ww; }
        else      { pix = s; }
        float u = xmc[((size_t)(b * Ln + pix)) * DIn + d];
        const float* drow = dblk + (size_t)s * 40;
        float dt = bdt;
#pragma unroll
        for (int r = 0; r < Rr; ++r) dt += wdt[r] * drow[r];
        dt = softplusf(dt);
        float Bv = drow[8  + n];
        float Cv = drow[24 + n];
        h = h * __expf(dt * Ac) + dt * u * Bv;
        float part = h * Cv;
        part += __shfl_xor(part, 1, 32);
        part += __shfl_xor(part, 2, 32);
        part += __shfl_xor(part, 4, 32);
        part += __shfl_xor(part, 8, 32);
        if (n == 0) {
            atomicAdd(&yc[((size_t)(b * Ln + pix)) * DIn + d], part + Dc * u);
        }
    }
}

// ---------------------------------------------------------------------------
// out_norm LayerNorm(DI) * silu(z) -> f16 rows (feeds out_proj GEMM only)
// ---------------------------------------------------------------------------
__global__ void gate_kernel(const float* __restrict__ yc, const float* __restrict__ xz,
                            const float* __restrict__ g, const float* __restrict__ be,
                            _Float16* __restrict__ yg16)
{
    __shared__ float red[DIn];
    int row = blockIdx.x;
    int d = threadIdx.x;
    float v = yc[(size_t)row * DIn + d];
    red[d] = v; __syncthreads();
    for (int s = DIn / 2; s > 0; s >>= 1) { if (d < s) red[d] += red[d + s]; __syncthreads(); }
    float mean = red[0] / (float)DIn; __syncthreads();
    float dv = v - mean; red[d] = dv * dv; __syncthreads();
    for (int s = DIn / 2; s > 0; s >>= 1) { if (d < s) red[d] += red[d + s]; __syncthreads(); }
    float var = red[0] / (float)DIn;
    float ln = dv * rsqrtf(var + 1e-5f) * g[d] + be[d];
    float z  = xz[(size_t)row * (2 * DIn) + DIn + d];
    yg16[(size_t)row * DIn + d] = (_Float16)(ln * siluf(z));
}

// LCF 3x3 conv (16->16), writes f16 rows padded to stride 32 (K-pad for GEMM)
__global__ void lcf_conv2_kernel(const float* __restrict__ h1, const float* __restrict__ w2,
                                 const float* __restrict__ b2, _Float16* __restrict__ h2)
{
    int idx = blockIdx.x * 256 + threadIdx.x;
    if (idx >= Bn * Ln * 16) return;
    int co = idx % 16; int bl = idx / 16;
    int b = bl / Ln; int l = bl - b * Ln;
    int hh = l / Wn; int ww = l - hh * Wn;
    float acc = b2[co];
    for (int ky = 0; ky < 3; ++ky) {
        int y = hh + ky - 1; if (y < 0 || y >= Hn) continue;
        for (int kx = 0; kx < 3; ++kx) {
            int x2 = ww + kx - 1; if (x2 < 0 || x2 >= Wn) continue;
            const float* src = h1 + ((size_t)(b * Ln + y * Wn + x2)) * 16;
            const float* wr  = w2 + (size_t)co * 144 + ky * 3 + kx;
#pragma unroll
            for (int ci = 0; ci < 16; ++ci) acc += wr[ci * 9] * src[ci];
        }
    }
    h2[(size_t)bl * 32 + co]      = (_Float16)acc;
    h2[(size_t)bl * 32 + 16 + co] = (_Float16)0.f;
}

// mean over H*W of h3 rows (B*L,C) -> p (B,C)
__global__ void pool_kernel(const float* __restrict__ h3, float* __restrict__ p)
{
    __shared__ float red[256];
    int bc = blockIdx.x; int b = bc / Cn; int c = bc - b * Cn;
    float acc = 0.f;
    for (int l = threadIdx.x; l < Ln; l += 256)
        acc += h3[((size_t)(b * Ln + l)) * Cn + c];
    red[threadIdx.x] = acc; __syncthreads();
    for (int s = 128; s > 0; s >>= 1) {
        if (threadIdx.x < s) red[threadIdx.x] += red[threadIdx.x + s];
        __syncthreads();
    }
    if (threadIdx.x == 0) p[bc] = red[0] / (float)Ln;
}

// s = sigmoid(relu(p @ f1^T) @ f2^T)  (tiny, one block)
__global__ void se_kernel(const float* __restrict__ p, const float* __restrict__ f1,
                          const float* __restrict__ f2, float* __restrict__ sg)
{
    __shared__ float q[16];
    int t = threadIdx.x;
    for (int b = 0; b < Bn; ++b) {
        if (t < 16) {
            float acc = 0.f;
            for (int c = 0; c < Cn; ++c) acc += p[b * Cn + c] * f1[t * Cn + c];
            q[t] = fmaxf(acc, 0.f);
        }
        __syncthreads();
        if (t < Cn) {
            float acc = 0.f;
#pragma unroll
            for (int j = 0; j < 16; ++j) acc += q[j] * f2[t * 16 + j];
            sg[b * Cn + t] = 1.f / (1.f + __expf(-acc));
        }
        __syncthreads();
    }
}

// out[b,c,h,w] = t2[b,l,c] + h3[b,l,c] * s[b,c]
__global__ void final_kernel(const float* __restrict__ t2, const float* __restrict__ h3,
                             const float* __restrict__ sg, float* __restrict__ out)
{
    int idx = blockIdx.x * 256 + threadIdx.x;
    if (idx >= Bn * Cn * Ln) return;
    int l = idx % Ln; int bc = idx / Ln;
    int b = bc / Cn; int c = bc - b * Cn;
    size_t r = ((size_t)(b * Ln + l)) * Cn + c;
    out[idx] = t2[r] + h3[r] * sg[bc];
}

// ---------------------------------------------------------------------------
extern "C" void kernel_launch(void* const* d_in, const int* in_sizes, int n_in,
                              void* d_out, int out_size, void* d_ws, size_t ws_size,
                              hipStream_t stream)
{
    (void)in_sizes; (void)n_in; (void)out_size; (void)ws_size;
    const float* x          = (const float*)d_in[0];
    const float* ln_in_g    = (const float*)d_in[1];
    const float* ln_in_b    = (const float*)d_in[2];
    const float* ln_ffn_g   = (const float*)d_in[3];
    const float* ln_ffn_b   = (const float*)d_in[4];
    const float* fc1_w      = (const float*)d_in[5];
    const float* fc1_b      = (const float*)d_in[6];
    const float* fc2_w      = (const float*)d_in[7];
    const float* fc2_b      = (const float*)d_in[8];
    const float* lcf_c1_w   = (const float*)d_in[9];
    const float* lcf_c1_b   = (const float*)d_in[10];
    const float* lcf_c2_w   = (const float*)d_in[11];
    const float* lcf_c2_b   = (const float*)d_in[12];
    const float* lcf_c3_w   = (const float*)d_in[13];
    const float* lcf_c3_b   = (const float*)d_in[14];
    const float* lcf_f1_w   = (const float*)d_in[15];
    const float* lcf_f2_w   = (const float*)d_in[16];
    const float* in_proj_w  = (const float*)d_in[17];
    const float* conv_w     = (const float*)d_in[18];
    const float* conv_b     = (const float*)d_in[19];
    const float* x_proj_w   = (const float*)d_in[20];
    const float* dt_w       = (const float*)d_in[21];
    const float* dt_b       = (const float*)d_in[22];
    const float* A_logs     = (const float*)d_in[23];
    const float* Ds         = (const float*)d_in[24];
    const float* out_norm_g = (const float*)d_in[25];
    const float* out_norm_b = (const float*)d_in[26];
    const float* out_proj_w = (const float*)d_in[27];

    float* ws = (float*)d_ws;
    size_t off = 0;
    auto allocf = [&](size_t n) { float* p = ws + off; off += n; return p; };

    float* xz   = allocf((size_t)Bn * Ln * 2 * DIn);      // f32 (B*L,512)
    float* xmc  = allocf((size_t)Bn * Ln * DIn);          // f32 (B*L,256)
    float* dblb = allocf((size_t)Bn * Kdir * Ln * 40);    // f32 (B,K,L,40)
    float* yc   = allocf((size_t)Bn * Ln * DIn);          // f32 scan accumulator
    float* xm2  = allocf((size_t)Bn * Ln * Cn);           // f32
    float* t2   = allocf((size_t)Bn * Ln * Cn);           // f32
    _Float16* xn16  = (_Float16*)allocf((size_t)Bn * Ln * Cn  / 2);
    _Float16* xmc16 = (_Float16*)allocf((size_t)Bn * Ln * DIn / 2);
    _Float16* yg16  = (_Float16*)allocf((size_t)Bn * Ln * DIn / 2);
    _Float16* ipw16 = (_Float16*)allocf((size_t)2 * DIn * Cn / 2);
    _Float16* opw16 = (_Float16*)allocf((size_t)Cn * DIn / 2);
    _Float16* f1w16 = (_Float16*)allocf((size_t)2 * Cn * Cn / 2);
    _Float16* f2w16 = (_Float16*)allocf((size_t)Cn * 2 * Cn / 2);
    _Float16* c1w16 = (_Float16*)allocf((size_t)16 * Cn / 2);
    _Float16* wpad  = (_Float16*)allocf((size_t)Kdir * 48 * DIn / 2);
    _Float16* c3p16 = (_Float16*)allocf((size_t)Cn * 32 / 2);
    // aliases (region dead by the time alias is written):
    _Float16* t0_16 = (_Float16*)yc;                       // after out_proj consumed yg16
    _Float16* t1_16 = (_Float16*)xmc;                      // after scan
    float*    h1b   = dblb;                                // after scan
    _Float16* h2b   = (_Float16*)(dblb + (size_t)Bn * Ln * 16);
    float*    pbuf  = dblb + (size_t)Bn * Ln * 16 * 2;     // h2 = Bn*Ln*32 halves
    float*    sbuf  = pbuf + Bn * Cn;
    float*    h3b   = xz;                                  // after gate consumed z

    const int M = Bn * Ln;                                 // 9216

    // ---- weight prep (f16 conversion / padding) ----
    cvt16_kernel<<<(2 * DIn * Cn + 255) / 256, 256, 0, stream>>>(in_proj_w, ipw16, 2 * DIn * Cn);
    cvt16_kernel<<<(Cn * DIn + 255) / 256, 256, 0, stream>>>(out_proj_w, opw16, Cn * DIn);
    cvt16_kernel<<<(2 * Cn * Cn + 255) / 256, 256, 0, stream>>>(fc1_w, f1w16, 2 * Cn * Cn);
    cvt16_kernel<<<(Cn * 2 * Cn + 255) / 256, 256, 0, stream>>>(fc2_w, f2w16, Cn * 2 * Cn);
    cvt16_kernel<<<(16 * Cn + 255) / 256, 256, 0, stream>>>(lcf_c1_w, c1w16, 16 * Cn);
    padw_kernel<<<(Kdir * 48 * DIn + 255) / 256, 256, 0, stream>>>(x_proj_w, wpad);
    padc3_kernel<<<(Cn * 32 + 255) / 256, 256, 0, stream>>>(lcf_c3_w, c3p16);

    // 1) input LayerNorm (eps 1e-6) -> f16 rows
    ln_bchw_kernel<<<M, Cn, 0, stream>>>(x, ln_in_g, ln_in_b, xn16, 1e-6f);
    // 2) in_proj: (9216x128)*(512x128)^T -> xz f32
    gemm_wmma_kernel<4><<<dim3(512 / 64, M / 64), 128, 0, stream>>>(
        xn16, ipw16, nullptr, nullptr, xz, nullptr, M, 512, Cn, 0, 0.f, 0, 0);
    // 3) depthwise 3x3 + SiLU -> xmc f32 + f16
    dwconv_kernel<<<(Bn * Ln * DIn + 255) / 256, 256, 0, stream>>>(xz, conv_w, conv_b, xmc, xmc16);
    // 4) x_proj per direction -> dbl
    xproj_wmma_kernel<<<dim3(1, Ln / 64, Bn * Kdir), 128, 0, stream>>>(xmc16, wpad, dblb);
    // 5) selective scan (all 4 directions accumulate into yc)
    zero_kernel<<<(Bn * Ln * DIn + 255) / 256, 256, 0, stream>>>(yc, Bn * Ln * DIn);
    scan_kernel<<<Bn * Kdir * (DIn / 2) / 4, 128, 0, stream>>>(xmc, dblb, dt_w, dt_b, A_logs, Ds, yc);
    // 6) out_norm LN * silu(z) -> yg16
    gate_kernel<<<M, DIn, 0, stream>>>(yc, xz, out_norm_g, out_norm_b, yg16);
    // LCF branch (xn16): 1x1 -> 3x3 -> 1x1 + leaky(0.2)
    gemm_wmma_kernel<1><<<dim3(1, M / 64), 128, 0, stream>>>(
        xn16, c1w16, lcf_c1_b, nullptr, h1b, nullptr, M, 16, Cn, 0, 0.f, 0, 0);
    lcf_conv2_kernel<<<(Bn * Ln * 16 + 255) / 256, 256, 0, stream>>>(h1b, lcf_c2_w, lcf_c2_b, h2b);
    gemm_wmma_kernel<4><<<dim3(2, M / 64), 128, 0, stream>>>(
        h2b, c3p16, lcf_c3_b, nullptr, h3b, nullptr, M, Cn, 32, 1, 0.2f, 0, 0);
    // 7) out_proj + residual(x, BCHW) -> xm2 f32
    gemm_wmma_kernel<4><<<dim3(2, M / 64), 128, 0, stream>>>(
        yg16, opw16, nullptr, x, xm2, nullptr, M, Cn, DIn, 0, 0.f, 2, 0);
    // 8) FFN: LN(1e-5) -> fc1 leaky(0.01) -> fc2 + residual
    ln_rows_kernel<<<M, Cn, 0, stream>>>(xm2, ln_ffn_g, ln_ffn_b, t0_16, 1e-5f);
    gemm_wmma_kernel<4><<<dim3(4, M / 64), 128, 0, stream>>>(
        t0_16, f1w16, fc1_b, nullptr, nullptr, t1_16, M, 2 * Cn, Cn, 1, 0.01f, 0, 1);
    gemm_wmma_kernel<4><<<dim3(2, M / 64), 128, 0, stream>>>(
        t1_16, f2w16, fc2_b, xm2, t2, nullptr, M, Cn, 2 * Cn, 0, 0.f, 1, 0);
    // 9) pool + SE gate + final fuse
    pool_kernel<<<Bn * Cn, 256, 0, stream>>>(h3b, pbuf);
    se_kernel<<<1, 128, 0, stream>>>(pbuf, lcf_f1_w, lcf_f2_w, sbuf);
    final_kernel<<<(Bn * Cn * Ln + 255) / 256, 256, 0, stream>>>(t2, h3b, sbuf, (float*)d_out);
}
